// TemporalMamba_75797582840221
// MI455X (gfx1250) — matrix operand order repaired and checked by reading, not compiled, not run on hardware
//
#include <hip/hip_runtime.h>

// ---------------------------------------------------------------------------
// TemporalMamba forward on MI455X (gfx1250, wave32, WMMA).
//
// Shapes: B=4, L=256, IN_DIM=16384, DM=512, DI=1024, DS=16, DC=4, DR=32, DEPTH=4.
// All GEMMs run on v_wmma_f32_16x16x32_bf16 (fp32 -> bf16 in-register, fp32 acc),
// one wave per 16x64 C tile (4 WMMA accumulators, A fragment reused 4x).
// The 16-row A slab shared by the 8 waves of a block is staged into LDS with
// global_load_async_to_lds_b128 (ASYNCcnt) in 256-wide K chunks: 8x less A
// traffic from L2 and one barrier pair amortized over 8 WMMA K-steps.
// Roofline: ~31 GFLOP total, ~160 MB HBM traffic (x + weights) -> memory-bound
// at 23.3 TB/s (~7us floor); activations live in the 192MB L2 between kernels.
// ---------------------------------------------------------------------------

#define B_     4
#define L_     256
#define INDIM_ 16384
#define DM_    512
#define DI_    1024
#define DS_    16
#define DC_    4
#define DR_    32
#define DEPTH_ 4
#define ML_    (B_ * L_)      // 1024 flattened (b,l) rows

#define KCHUNK 256            // K columns staged in LDS per barrier pair
#define ASTRIDE 260           // LDS row stride (dwords): 4-dword shift/row -> bank-spread

typedef __attribute__((ext_vector_type(16))) __bf16 v16bf;
typedef __attribute__((ext_vector_type(8)))  float  v8f;

__device__ __forceinline__ float softplus_f(float x) {
  return (x > 20.f) ? x : log1pf(__expf(x));
}
__device__ __forceinline__ float silu_f(float x) {
  return x / (1.f + __expf(-x));
}

// Convert 16 contiguous fp32 (16B-aligned) to a bf16 WMMA fragment half-row.
__device__ __forceinline__ v16bf cvt_frag16(const float* __restrict__ p) {
  float f[16];
  *(float4*)(f + 0)  = *(const float4*)(p + 0);
  *(float4*)(f + 4)  = *(const float4*)(p + 4);
  *(float4*)(f + 8)  = *(const float4*)(p + 8);
  *(float4*)(f + 12) = *(const float4*)(p + 12);
  v16bf r;
#pragma unroll
  for (int i = 0; i < 16; ++i) r[i] = (__bf16)f[i];
  return r;
}

// Build bf16 A fragment (ISA 7.12.2 16-bit A 16x32 layout) from a row pointer:
// e0..7 = K[ka..ka+7], e8..15 = K[ka+16..ka+23]; ka already includes half*8.
__device__ __forceinline__ v16bf cvt_afrag(const float* __restrict__ p) {
  float f[16];
  *(float4*)(f + 0)  = *(const float4*)(p + 0);
  *(float4*)(f + 4)  = *(const float4*)(p + 4);
  *(float4*)(f + 8)  = *(const float4*)(p + 16);
  *(float4*)(f + 12) = *(const float4*)(p + 20);
  v16bf r;
#pragma unroll
  for (int i = 0; i < 16; ++i) r[i] = (__bf16)f[i];
  return r;
}

// ---------------------------------------------------------------------------
// Generic GEMM: C[M,N] = A[M,K] * W[N,K]^T (+ bias[N]) (+ softplus if mode==1)
// A row-major (row stride lda); W row-major (row stride ldw) so the K-dot walks
// contiguous memory in BOTH operands; C row stride ldc.
// One wave per 16x64 C tile (4 x v_wmma_f32_16x16x32_bf16 per K step),
// 8 waves / 256-thread block.
// staged=1 (requires (N/64)%8==0): block's shared 16-row A slab is copied to
// LDS via global_load_async_to_lds_b128 in KCHUNK-wide pieces.
// Requires: N % 64 == 0, K % 32 == 0, lda/ldw % 4 == 0. M may be ragged:
// out-of-range A rows are clamped to row 0 (their C rows are never stored).
// ---------------------------------------------------------------------------
__global__ __launch_bounds__(256) void gemm_wmma_bf16(
    const float* __restrict__ A, int lda,
    const float* __restrict__ W, int ldw,
    const float* __restrict__ bias,
    float* __restrict__ C, int ldc,
    int M, int N, int K, int mode, int staged)
{
  __shared__ float atile[16 * ASTRIDE];            // ~16.6 KB staged A slab

  const int wave = threadIdx.x >> 5;
  const int lane = threadIdx.x & 31;
  const int ntn  = N >> 6;                         // 64-wide N tiles
  const int ntm  = (M + 15) >> 4;
  const int tile = blockIdx.x * 8 + wave;          // wave-uniform
  if (tile >= ntm * ntn) return;                   // only possible when staged==0
  const int tm = tile / ntn;
  const int tn = tile % ntn;

  const int half = lane >> 4;                      // 0: lanes 0-15, 1: lanes 16-31
  const int lr   = lane & 15;

  // A row owned by this lane; clamp OOB rows to 0 (results discarded at store).
  int am = tm * 16 + lr;
  if (am >= M) am = 0;
  const float* __restrict__ Arow = A + (size_t)am * lda;
  // Four B columns per lane (one per 16-wide sub-tile).
  const float* __restrict__ W0 = W + (size_t)(tn * 64 + 0 * 16 + lr) * ldw;
  const float* __restrict__ W1 = W + (size_t)(tn * 64 + 1 * 16 + lr) * ldw;
  const float* __restrict__ W2 = W + (size_t)(tn * 64 + 2 * 16 + lr) * ldw;
  const float* __restrict__ W3 = W + (size_t)(tn * 64 + 3 * 16 + lr) * ldw;

  v8f acc0 = {}, acc1 = {}, acc2 = {}, acc3 = {};

  if (staged) {
    // LDS aperture: low 32 bits of a generic shared pointer == LDS byte offset.
    const unsigned ldsbase = (unsigned)(uintptr_t)(void*)atile;
    // Cooperative copy mapping: 16 threads per A row.
    const int crow  = threadIdx.x >> 4;            // 0..15
    const int cbase = (threadIdx.x & 15) * 4;      // 0..60
    int grow = tm * 16 + crow;
    if (grow >= M) grow = 0;
    const float* __restrict__ gsrc = A + (size_t)grow * lda;

    for (int kc = 0; kc < K; kc += KCHUNK) {
      const int chunk = (K - kc < KCHUNK) ? (K - kc) : KCHUNK;

      // Async global -> LDS copy of A[tm*16..+16)[kc..kc+chunk) (fp32, b128/lane).
      for (int c = cbase; c < chunk; c += 64) {
        const unsigned dst = ldsbase + (unsigned)((crow * ASTRIDE + c) * 4);
        const unsigned long long ga = (unsigned long long)(uintptr_t)(gsrc + kc + c);
        asm volatile("global_load_async_to_lds_b128 %0, %1, off"
                     :: "v"(dst), "v"(ga) : "memory");
      }
      asm volatile("s_wait_asynccnt 0x0" ::: "memory");  // this wave's copies done
      __syncthreads();                                    // publish all waves' copies

      for (int kk = 0; kk < chunk; kk += 32) {
        const float* ap = atile + lr * ASTRIDE + kk + half * 16;
        v16bf a  = cvt_frag16(ap);                        // ds_load_b128 x4 + cvt
        const int kb = kc + kk + half * 16;
        __builtin_prefetch(W0 + kb + 32, 0, 1);
        __builtin_prefetch(W2 + kb + 32, 0, 1);
        v16bf b0 = cvt_frag16(W0 + kb);
        v16bf b1 = cvt_frag16(W1 + kb);
        v16bf b2 = cvt_frag16(W2 + kb);
        v16bf b3 = cvt_frag16(W3 + kb);
        acc0 = __builtin_amdgcn_wmma_f32_16x16x32_bf16(false, a, false, b0, (short)0, acc0, false, false);
        acc1 = __builtin_amdgcn_wmma_f32_16x16x32_bf16(false, a, false, b1, (short)0, acc1, false, false);
        acc2 = __builtin_amdgcn_wmma_f32_16x16x32_bf16(false, a, false, b2, (short)0, acc2, false, false);
        acc3 = __builtin_amdgcn_wmma_f32_16x16x32_bf16(false, a, false, b3, (short)0, acc3, false, false);
      }
      __syncthreads();                                    // safe to overwrite slab
    }
  } else {
    // Direct path (used when the block's waves do not share an A slab, N=64).
    for (int k0 = 0; k0 < K; k0 += 32) {
      __builtin_prefetch(Arow + k0 + 32, 0, 1);
      __builtin_prefetch(W0 + k0 + 32, 0, 1);
      __builtin_prefetch(W2 + k0 + 32, 0, 1);
      v16bf a  = cvt_frag16(Arow + k0 + half * 16);
      const int kb = k0 + half * 16;
      v16bf b0 = cvt_frag16(W0 + kb);
      v16bf b1 = cvt_frag16(W1 + kb);
      v16bf b2 = cvt_frag16(W2 + kb);
      v16bf b3 = cvt_frag16(W3 + kb);
      acc0 = __builtin_amdgcn_wmma_f32_16x16x32_bf16(false, a, false, b0, (short)0, acc0, false, false);
      acc1 = __builtin_amdgcn_wmma_f32_16x16x32_bf16(false, a, false, b1, (short)0, acc1, false, false);
      acc2 = __builtin_amdgcn_wmma_f32_16x16x32_bf16(false, a, false, b2, (short)0, acc2, false, false);
      acc3 = __builtin_amdgcn_wmma_f32_16x16x32_bf16(false, a, false, b3, (short)0, acc3, false, false);
    }
  }

  // Epilogue. C layout: VGPR r -> row tm*16 + r + 8*half, col = sub*16 + lr.
  const int cn0 = tn * 64 + lr;
  float bv0 = bias ? bias[cn0 + 0]  : 0.f;
  float bv1 = bias ? bias[cn0 + 16] : 0.f;
  float bv2 = bias ? bias[cn0 + 32] : 0.f;
  float bv3 = bias ? bias[cn0 + 48] : 0.f;

  const bool fullM = (tm * 16 + 16 <= M);
#pragma unroll
  for (int r = 0; r < 8; ++r) {
    const int cm = tm * 16 + r + half * 8;
    if (fullM || cm < M) {
      float* Crow = C + (size_t)cm * ldc + cn0;
      float v0 = acc0[r] + bv0;
      float v1 = acc1[r] + bv1;
      float v2 = acc2[r] + bv2;
      float v3 = acc3[r] + bv3;
      if (mode == 1) {
        v0 = softplus_f(v0); v1 = softplus_f(v1);
        v2 = softplus_f(v2); v3 = softplus_f(v3);
      }
      Crow[0]  = v0;
      Crow[16] = v1;
      Crow[32] = v2;
      Crow[48] = v3;
    }
  }
}

// ---------------------------------------------------------------------------
// Causal depthwise conv (DC=4 taps) over L on the xc half of xz, + bias, + silu.
// ---------------------------------------------------------------------------
__global__ void conv_silu_kernel(const float* __restrict__ xz,
                                 const float* __restrict__ cw,   // [DI,DC]
                                 const float* __restrict__ cb,   // [DI]
                                 float* __restrict__ xc)
{
  int idx = blockIdx.x * blockDim.x + threadIdx.x;   // over ML_*DI_
  if (idx >= ML_ * DI_) return;
  const int di = idx % DI_;
  const int bl = idx / DI_;
  const int l  = bl % L_;
  const int b  = bl / L_;
  float s = cb[di];
#pragma unroll
  for (int k = 0; k < DC_; ++k) {
    const int ls = l - (DC_ - 1) + k;
    if (ls >= 0)
      s += xz[((size_t)(b * L_ + ls)) * (2 * DI_) + di] * cw[di * DC_ + k];
  }
  xc[idx] = silu_f(s);
}

// ---------------------------------------------------------------------------
// Selective-scan: one thread per (b, di); DS=16 states in registers; L=256
// sequential steps. Fuses the D skip-path and the silu(z) gate.
// y[b,l,di] = (sum_s h_s * C_s + xc*Dp) * silu(z)
// ---------------------------------------------------------------------------
__global__ void ssm_scan_kernel(const float* __restrict__ xc,    // [ML,DI]
                                const float* __restrict__ xdbl,  // [ML,64] = dt|B|C
                                const float* __restrict__ dt,    // [ML,DI]
                                const float* __restrict__ xz,    // [ML,2DI] (z half)
                                const float* __restrict__ Alog,  // [DI,DS]
                                const float* __restrict__ Dp,    // [DI]
                                float* __restrict__ y)           // [ML,DI]
{
  int idx = blockIdx.x * blockDim.x + threadIdx.x;   // over B_*DI_
  if (idx >= B_ * DI_) return;
  const int di = idx % DI_;
  const int b  = idx / DI_;

  float A[DS_];
#pragma unroll
  for (int s = 0; s < DS_; ++s) A[s] = -__expf(Alog[di * DS_ + s]);
  float hst[DS_];
#pragma unroll
  for (int s = 0; s < DS_; ++s) hst[s] = 0.f;
  const float dp = Dp[di];

  for (int l = 0; l < L_; ++l) {
    const size_t bl  = (size_t)b * L_ + l;
    const float  dtv = dt[bl * DI_ + di];
    const float  xv  = xc[bl * DI_ + di];
    const float* xd  = xdbl + bl * 64;               // [0,32)=dt, [32,48)=B, [48,64)=C
    float Bm[DS_], Cm[DS_];
    *(float4*)(Bm + 0)  = *(const float4*)(xd + DR_ + 0);
    *(float4*)(Bm + 4)  = *(const float4*)(xd + DR_ + 4);
    *(float4*)(Bm + 8)  = *(const float4*)(xd + DR_ + 8);
    *(float4*)(Bm + 12) = *(const float4*)(xd + DR_ + 12);
    *(float4*)(Cm + 0)  = *(const float4*)(xd + DR_ + DS_ + 0);
    *(float4*)(Cm + 4)  = *(const float4*)(xd + DR_ + DS_ + 4);
    *(float4*)(Cm + 8)  = *(const float4*)(xd + DR_ + DS_ + 8);
    *(float4*)(Cm + 12) = *(const float4*)(xd + DR_ + DS_ + 12);
    float yv = 0.f;
#pragma unroll
    for (int s = 0; s < DS_; ++s) {
      const float dA = __expf(dtv * A[s]);
      hst[s] = dA * hst[s] + dtv * Bm[s] * xv;
      yv += hst[s] * Cm[s];
    }
    yv += xv * dp;
    const float zv = xz[bl * (2 * DI_) + DI_ + di];
    y[bl * DI_ + di] = yv * silu_f(zv);
  }
}

// ---------------------------------------------------------------------------
// LayerNorm of the last token per batch: hn[b,:] = LN(h[b,L-1,:])*g + beta
// One 512-thread block per batch row.
// ---------------------------------------------------------------------------
__global__ void ln_last_kernel(const float* __restrict__ h,
                               const float* __restrict__ g,
                               const float* __restrict__ be,
                               float* __restrict__ hn)
{
  __shared__ float sh[DM_];
  __shared__ float sh2[DM_];
  const int b = blockIdx.x;
  const int t = threadIdx.x;
  const float v = h[((size_t)b * L_ + (L_ - 1)) * DM_ + t];
  sh[t] = v; sh2[t] = v * v;
  __syncthreads();
  for (int off = DM_ / 2; off > 0; off >>= 1) {
    if (t < off) { sh[t] += sh[t + off]; sh2[t] += sh2[t + off]; }
    __syncthreads();
  }
  const float mu  = sh[0] * (1.f / DM_);
  const float var = sh2[0] * (1.f / DM_) - mu * mu;
  const float r   = rsqrtf(var + 1e-5f);
  hn[b * DM_ + t] = (v - mu) * r * g[t] + be[t];
}

// ---------------------------------------------------------------------------
// Host-side orchestration (all launches on `stream`, graph-capture safe).
// ---------------------------------------------------------------------------
extern "C" void kernel_launch(void* const* d_in, const int* in_sizes, int n_in,
                              void* d_out, int out_size, void* d_ws, size_t ws_size,
                              hipStream_t stream) {
  const float* x       = (const float*)d_in[0];
  const float* in_w    = (const float*)d_in[1];
  const float* in_b    = (const float*)d_in[2];
  const float* ln_g    = (const float*)d_in[3];
  const float* ln_b    = (const float*)d_in[4];
  const float* out_w   = (const float*)d_in[5];
  const float* out_b   = (const float*)d_in[6];
  const float* m_in_w  = (const float*)d_in[7];
  const float* conv_w  = (const float*)d_in[8];
  const float* conv_b  = (const float*)d_in[9];
  const float* xproj_w = (const float*)d_in[10];
  const float* dt_w    = (const float*)d_in[11];
  const float* dt_b    = (const float*)d_in[12];
  const float* A_log   = (const float*)d_in[13];
  const float* D_p     = (const float*)d_in[14];
  const float* m_out_w = (const float*)d_in[15];
  float* out = (float*)d_out;

  // Workspace carve-up (floats): every buffer fully written before read.
  float* ws   = (float*)d_ws;
  float* h    = ws;                                  // ML * DM
  float* xzb  = h    + (size_t)ML_ * DM_;            // ML * 2DI
  float* xcb  = xzb  + (size_t)ML_ * 2 * DI_;        // ML * DI
  float* xdbl = xcb  + (size_t)ML_ * DI_;            // ML * 64
  float* dtb  = xdbl + (size_t)ML_ * 64;             // ML * DI
  float* yb   = dtb  + (size_t)ML_ * DI_;            // ML * DI
  float* hn   = yb   + (size_t)ML_ * DI_;            // B * DM

  auto gemm = [&](const float* A, int lda, const float* W, int ldw,
                  const float* bias, float* C, int ldc,
                  int M, int N, int K, int mode) {
    const int ntn    = N / 64;
    const int staged = (ntn % 8 == 0) ? 1 : 0;       // all 8 waves share the A slab
    const int tiles  = ((M + 15) / 16) * ntn;
    const int blocks = (tiles + 7) / 8;              // staged grids are always full
    gemm_wmma_bf16<<<blocks, 256, 0, stream>>>(A, lda, W, ldw, bias, C, ldc,
                                               M, N, K, mode, staged);
  };

  // Input projection: h = x @ in_w.T + in_b   [ML, DM]
  gemm(x, INDIM_, in_w, INDIM_, in_b, h, DM_, ML_, DM_, INDIM_, 0);

  for (int d = 0; d < DEPTH_; ++d) {
    const float* iw = m_in_w  + (size_t)d * 2 * DI_ * DM_;   // [2DI, DM]
    const float* cw = conv_w  + (size_t)d * DI_ * DC_;
    const float* cb = conv_b  + (size_t)d * DI_;
    const float* xw = xproj_w + (size_t)d * (DR_ + 2 * DS_) * DI_;  // [64, DI]
    const float* dw = dt_w    + (size_t)d * DI_ * DR_;       // [DI, DR]
    const float* db = dt_b    + (size_t)d * DI_;
    const float* Al = A_log   + (size_t)d * DI_ * DS_;
    const float* Dd = D_p     + (size_t)d * DI_;
    const float* ow = m_out_w + (size_t)d * DM_ * DI_;       // [DM, DI]

    // xz = h @ iw.T                              [ML, 2DI]
    gemm(h, DM_, iw, DM_, nullptr, xzb, 2 * DI_, ML_, 2 * DI_, DM_, 0);
    // xc = silu(causal_conv(xz[..., :DI]))       [ML, DI]
    conv_silu_kernel<<<(ML_ * DI_ + 255) / 256, 256, 0, stream>>>(xzb, cw, cb, xcb);
    // xdbl = xc @ xw.T                           [ML, 64]  (direct path, N=64)
    gemm(xcb, DI_, xw, DI_, nullptr, xdbl, 64, ML_, 64, DI_, 0);
    // dt = softplus(xdbl[:, :DR] @ dw.T + db)    [ML, DI]  (K=32, lda=64)
    gemm(xdbl, 64, dw, DR_, db, dtb, DI_, ML_, DI_, DR_, 1);
    // selective scan (fuses D skip + silu(z) gate) -> y  [ML, DI]
    ssm_scan_kernel<<<(B_ * DI_ + 255) / 256, 256, 0, stream>>>(
        xcb, xdbl, dtb, xzb, Al, Dd, yb);
    // h = y @ ow.T                               [ML, DM]
    gemm(yb, DI_, ow, DI_, nullptr, h, DM_, ML_, DM_, DI_, 0);
  }

  // LayerNorm of last token, then output head: out = hn @ out_w.T + out_b
  ln_last_kernel<<<B_, DM_, 0, stream>>>(h, ln_g, ln_b, hn);
  gemm(hn, DM_, out_w, DM_, out_b, out, INDIM_, B_, INDIM_, DM_, 0);
}
